// MaskedMultiHeadAttention_11003706213255
// MI455X (gfx1250) — matrix operand order repaired and checked by reading, not compiled, not run on hardware
//
#include <hip/hip_runtime.h>

typedef __attribute__((ext_vector_type(16))) _Float16 v16h;
typedef __attribute__((ext_vector_type(8)))  float    v8f;

#define BQ   4
#define SQ   2048
#define DMOD 1024
#define NH   16
#define HD   64
#define MTOT 8192   // B*S

static __device__ __forceinline__ v8f wmma32(v16h a, v16h b, v8f c) {
  // D = A(16x32 f16) * B(32x16 f16) + C(16x16 f32)
  return __builtin_amdgcn_wmma_f32_16x16x32_f16(
      false, a, false, b, (short)0, c, false, false);
}

// ---- A fragment (16x32, f16) from row-major f16 source, leading dim ld.
// ISA layout: lanes 0-15 -> row M=lane, K pairs {0..7,16..23}; lanes 16-31 -> +8.
static __device__ __forceinline__ v16h load_a_f16(const _Float16* base, int ld) {
  int lane = threadIdx.x & 31;
  int row  = lane & 15;
  int hh   = lane >> 4;
  const _Float16* p = base + row * ld + 8 * hh;
  v16h a;
#pragma unroll
  for (int i = 0; i < 8; ++i) {
    int k = (i < 4) ? (2 * i) : (2 * i + 8);
    a[2 * i]     = p[k];
    a[2 * i + 1] = p[k + 1];
  }
  return a;
}

// ---- A fragment from fp32 source with convert-to-f16 on the fly.
static __device__ __forceinline__ v16h load_a_f32(const float* base, int ld) {
  int lane = threadIdx.x & 31;
  int row  = lane & 15;
  int hh   = lane >> 4;
  const float* p = base + row * ld + 8 * hh;
  v16h a;
#pragma unroll
  for (int i = 0; i < 8; ++i) {
    int k = (i < 4) ? (2 * i) : (2 * i + 8);
    a[2 * i]     = (_Float16)p[k];
    a[2 * i + 1] = (_Float16)p[k + 1];
  }
  return a;
}

// ---- B fragment (32x16, f16) from "B-transposed" storage bt[n][k] (ld = ldb).
// ISA layout: lane n = lane&15, K {0..7,16..23} for lanes 0-15; +8 for lanes 16-31.
static __device__ __forceinline__ v16h load_bt(const _Float16* bt, int ldb) {
  int lane = threadIdx.x & 31;
  int n  = lane & 15;
  int k0 = (lane >> 4) * 8;
  const _Float16* p = bt + n * ldb + k0;
  v16h b;
#pragma unroll
  for (int s = 0; s < 8; ++s) {
    b[s]     = p[s];
    b[8 + s] = p[16 + s];
  }
  return b;
}

// ============================ kernel 1: weight prep =========================
// wt[n*1024 + k] = (f16) w[k*1024 + n]
__global__ void k_transpose_w(const float* __restrict__ w, _Float16* __restrict__ wt) {
  int idx = blockIdx.x * blockDim.x + threadIdx.x;
  int k = idx >> 10;
  int n = idx & 1023;
  wt[n * 1024 + k] = (_Float16)w[k * 1024 + n];
}

// ============================ kernel 2: QKV GEMM ============================
// x:[8192,1024] f32 ; wt:[1024,1024] f16 (transposed) ; out head-split [B,H,S,64] f16
// Block: 256 threads = 8 waves, tile 256x64; wave tile 32x64 (2 A frags share 4 B frags).
__global__ void k_qkv_gemm(const float* __restrict__ x, const _Float16* __restrict__ wt,
                           const float* __restrict__ bias, _Float16* __restrict__ out_bhsd) {
  int w    = threadIdx.x >> 5;
  int lane = threadIdx.x & 31;
  int m0 = blockIdx.x * 256 + w * 32;
  int n0 = blockIdx.y * 64;

  v8f acc[2][4];
#pragma unroll
  for (int i = 0; i < 2; ++i)
#pragma unroll
    for (int j = 0; j < 4; ++j) acc[i][j] = v8f{};

  for (int k0 = 0; k0 < 1024; k0 += 32) {
    v16h a0 = load_a_f32(x + (size_t)m0 * 1024 + k0, 1024);
    v16h a1 = load_a_f32(x + (size_t)(m0 + 16) * 1024 + k0, 1024);
    v16h b0 = load_bt(wt + (size_t)(n0 +  0) * 1024 + k0, 1024);
    v16h b1 = load_bt(wt + (size_t)(n0 + 16) * 1024 + k0, 1024);
    v16h b2 = load_bt(wt + (size_t)(n0 + 32) * 1024 + k0, 1024);
    v16h b3 = load_bt(wt + (size_t)(n0 + 48) * 1024 + k0, 1024);
    acc[0][0] = wmma32(a0, b0, acc[0][0]);
    acc[1][0] = wmma32(a1, b0, acc[1][0]);
    acc[0][1] = wmma32(a0, b1, acc[0][1]);
    acc[1][1] = wmma32(a1, b1, acc[1][1]);
    acc[0][2] = wmma32(a0, b2, acc[0][2]);
    acc[1][2] = wmma32(a1, b2, acc[1][2]);
    acc[0][3] = wmma32(a0, b3, acc[0][3]);
    acc[1][3] = wmma32(a1, b3, acc[1][3]);
  }

  int col  = lane & 15;
  int rowh = (lane >> 4) * 8;
#pragma unroll
  for (int j = 0; j < 4; ++j) {
    int c = n0 + j * 16 + col;
    float bv = bias[c];
    int h  = c >> 6;
    int dk = c & 63;
#pragma unroll
    for (int i = 0; i < 2; ++i) {
#pragma unroll
      for (int r = 0; r < 8; ++r) {
        int m = m0 + i * 16 + rowh + r;
        int b = m >> 11;
        int s = m & 2047;
        out_bhsd[((size_t)(b * NH + h) * SQ + s) * HD + dk] = (_Float16)(acc[i][j][r] + bv);
      }
    }
  }
}

// ========================= kernel 3: flash attention ========================
// Q,K,V: [B,H,S,64] f16.  O: [B*S, 1024] f16 (head-concat).  mask: [B,S] int.
__global__ void k_attn(const _Float16* __restrict__ Q, const _Float16* __restrict__ K,
                       const _Float16* __restrict__ V, const int* __restrict__ mask,
                       _Float16* __restrict__ O) {
  __shared__ __align__(64) _Float16 Ksh[128][64];       // K tile, row-major [key][dk]
  __shared__ __align__(64) _Float16 VT[64][128];        // V tile transposed [dv][key]
  __shared__ __align__(64) _Float16 P[8][16][128];      // per-wave P tile

  int w    = threadIdx.x >> 5;
  int lane = threadIdx.x & 31;
  int bh = blockIdx.y;
  int b  = bh >> 4;
  int h  = bh & 15;
  int q0 = blockIdx.x * 128 + w * 16;

  const _Float16* Qp = Q + (size_t)bh * SQ * HD;
  const _Float16* Kp = K + (size_t)bh * SQ * HD;
  const _Float16* Vp = V + (size_t)bh * SQ * HD;
  const int* mp = mask + b * SQ;

  v16h aq0 = load_a_f16(Qp + (size_t)q0 * HD + 0, HD);
  v16h aq1 = load_a_f16(Qp + (size_t)q0 * HD + 32, HD);

  float mrun[8], lrun[8];
  v8f o[4];
#pragma unroll
  for (int r = 0; r < 8; ++r) { mrun[r] = -3.0e38f; lrun[r] = 0.0f; }
#pragma unroll
  for (int j = 0; j < 4; ++j) o[j] = v8f{};

  int prow = (lane >> 4) * 8;
  unsigned kshBase = (unsigned)(size_t)(&Ksh[0][0]);   // wave-relative LDS byte offset

  for (int kt = 0; kt < SQ / 128; ++kt) {
    int kb = kt * 128;

    __syncthreads();   // previous tile fully consumed

    // ---- async copy K tile (16KB) global -> LDS, 64B per thread
    {
      unsigned ldsb = kshBase + threadIdx.x * 64u;
      const _Float16* g = Kp + (size_t)kb * HD + threadIdx.x * 32;
      asm volatile("global_load_async_to_lds_b128 %0, %1, off" :: "v"(ldsb),       "v"(g)      : "memory");
      asm volatile("global_load_async_to_lds_b128 %0, %1, off" :: "v"(ldsb + 16u), "v"(g + 8)  : "memory");
      asm volatile("global_load_async_to_lds_b128 %0, %1, off" :: "v"(ldsb + 32u), "v"(g + 16) : "memory");
      asm volatile("global_load_async_to_lds_b128 %0, %1, off" :: "v"(ldsb + 48u), "v"(g + 24) : "memory");
    }

    // ---- stage V tile transposed into LDS: VT[dv][key]
    for (int idx = threadIdx.x; idx < 128 * 16; idx += 256) {
      int key = idx >> 4;
      int d4  = (idx & 15) * 4;
      const _Float16* src = Vp + (size_t)(kb + key) * HD + d4;
      VT[d4 + 0][key] = src[0];
      VT[d4 + 1][key] = src[1];
      VT[d4 + 2][key] = src[2];
      VT[d4 + 3][key] = src[3];
    }
    asm volatile("s_wait_asynccnt 0x0" ::: "memory");
    __syncthreads();

    if (kb + 128 < SQ)  // prefetch next K tile -> global_prefetch
      __builtin_prefetch(Kp + (size_t)(kb + 128) * HD, 0, 0);

    // ---- logits S = Q K^T * 0.125 + mask  (K fragments from LDS)
    v8f sc[8];
    float tmax[8];
#pragma unroll
    for (int r = 0; r < 8; ++r) tmax[r] = -3.0e38f;
#pragma unroll
    for (int ns = 0; ns < 8; ++ns) {
      v8f c = v8f{};
      v16h bk0 = load_bt(&Ksh[ns * 16][0] + 0, 64);
      v16h bk1 = load_bt(&Ksh[ns * 16][0] + 32, 64);
      c = wmma32(aq0, bk0, c);
      c = wmma32(aq1, bk1, c);
      int key = kb + ns * 16 + (lane & 15);
      float madd = (float)mp[key] * -1.0e9f;
#pragma unroll
      for (int r = 0; r < 8; ++r) {
        float s = c[r] * 0.125f + madd;
        sc[ns][r] = s;
        tmax[r] = fmaxf(tmax[r], s);
      }
    }
    // row max across the 16 lanes holding each row
#pragma unroll
    for (int r = 0; r < 8; ++r) {
      float t = tmax[r];
      t = fmaxf(t, __shfl_xor(t, 1, 16));
      t = fmaxf(t, __shfl_xor(t, 2, 16));
      t = fmaxf(t, __shfl_xor(t, 4, 16));
      t = fmaxf(t, __shfl_xor(t, 8, 16));
      tmax[r] = t;
    }
    float corr[8];
#pragma unroll
    for (int r = 0; r < 8; ++r) {
      float nm = fmaxf(mrun[r], tmax[r]);
      corr[r] = __expf(mrun[r] - nm);
      mrun[r] = nm;
    }
    // P = exp(S - m), write to per-wave LDS in A-matrix storage order
    float tsum[8] = {0.f, 0.f, 0.f, 0.f, 0.f, 0.f, 0.f, 0.f};
#pragma unroll
    for (int ns = 0; ns < 8; ++ns) {
#pragma unroll
      for (int r = 0; r < 8; ++r) {
        float p = __expf(sc[ns][r] - mrun[r]);
        tsum[r] += p;
        P[w][prow + r][ns * 16 + (lane & 15)] = (_Float16)p;
      }
    }
    asm volatile("s_wait_dscnt 0x0" ::: "memory");  // wave-local LDS RAW
#pragma unroll
    for (int r = 0; r < 8; ++r) {
      float t = tsum[r];
      t += __shfl_xor(t, 1, 16);
      t += __shfl_xor(t, 2, 16);
      t += __shfl_xor(t, 4, 16);
      t += __shfl_xor(t, 8, 16);
      lrun[r] = lrun[r] * corr[r] + t;
    }
#pragma unroll
    for (int j = 0; j < 4; ++j)
#pragma unroll
      for (int r = 0; r < 8; ++r) o[j][r] *= corr[r];

    // ---- O += P x V
    v16h ap[4];
#pragma unroll
    for (int kk = 0; kk < 4; ++kk)
      ap[kk] = load_a_f16(&P[w][0][0] + kk * 32, 128);
#pragma unroll
    for (int j = 0; j < 4; ++j) {
#pragma unroll
      for (int kk = 0; kk < 4; ++kk) {
        v16h bv = load_bt(&VT[j * 16][0] + kk * 32, 128);
        o[j] = wmma32(ap[kk], bv, o[j]);
      }
    }
  }

  // epilogue: O / l, store f16 head-concat [B*S, 1024]
  float inv[8];
#pragma unroll
  for (int r = 0; r < 8; ++r) inv[r] = 1.0f / lrun[r];
#pragma unroll
  for (int j = 0; j < 4; ++j) {
    int col = h * HD + j * 16 + (lane & 15);
#pragma unroll
    for (int r = 0; r < 8; ++r) {
      int row = b * SQ + q0 + prow + r;
      O[(size_t)row * 1024 + col] = (_Float16)(o[j][r] * inv[r]);
    }
  }
}

// ======================== kernel 4: output projection =======================
// x:[8192,1024] f16 ; wt:[1024,1024] f16 (transposed) ; out fp32 + bias
__global__ void k_out_gemm(const _Float16* __restrict__ x, const _Float16* __restrict__ wt,
                           const float* __restrict__ bias, float* __restrict__ out) {
  int w    = threadIdx.x >> 5;
  int lane = threadIdx.x & 31;
  int m0 = blockIdx.x * 256 + w * 32;
  int n0 = blockIdx.y * 64;

  v8f acc[2][4];
#pragma unroll
  for (int i = 0; i < 2; ++i)
#pragma unroll
    for (int j = 0; j < 4; ++j) acc[i][j] = v8f{};

  for (int k0 = 0; k0 < 1024; k0 += 32) {
    v16h a0 = load_a_f16(x + (size_t)m0 * 1024 + k0, 1024);
    v16h a1 = load_a_f16(x + (size_t)(m0 + 16) * 1024 + k0, 1024);
    v16h b0 = load_bt(wt + (size_t)(n0 +  0) * 1024 + k0, 1024);
    v16h b1 = load_bt(wt + (size_t)(n0 + 16) * 1024 + k0, 1024);
    v16h b2 = load_bt(wt + (size_t)(n0 + 32) * 1024 + k0, 1024);
    v16h b3 = load_bt(wt + (size_t)(n0 + 48) * 1024 + k0, 1024);
    acc[0][0] = wmma32(a0, b0, acc[0][0]);
    acc[1][0] = wmma32(a1, b0, acc[1][0]);
    acc[0][1] = wmma32(a0, b1, acc[0][1]);
    acc[1][1] = wmma32(a1, b1, acc[1][1]);
    acc[0][2] = wmma32(a0, b2, acc[0][2]);
    acc[1][2] = wmma32(a1, b2, acc[1][2]);
    acc[0][3] = wmma32(a0, b3, acc[0][3]);
    acc[1][3] = wmma32(a1, b3, acc[1][3]);
  }

  int col  = lane & 15;
  int rowh = (lane >> 4) * 8;
#pragma unroll
  for (int j = 0; j < 4; ++j) {
    int c = n0 + j * 16 + col;
    float bv = bias[c];
#pragma unroll
    for (int i = 0; i < 2; ++i) {
#pragma unroll
      for (int r = 0; r < 8; ++r) {
        int m = m0 + i * 16 + rowh + r;
        out[(size_t)m * 1024 + c] = acc[i][j][r] + bv;
      }
    }
  }
}

// ================================ launcher ==================================
extern "C" void kernel_launch(void* const* d_in, const int* in_sizes, int n_in,
                              void* d_out, int out_size, void* d_ws, size_t ws_size,
                              hipStream_t stream) {
  const float* query = (const float*)d_in[0];
  const float* key   = (const float*)d_in[1];
  const float* value = (const float*)d_in[2];
  const int*   amask = (const int*)d_in[3];
  const float* wq = (const float*)d_in[4];
  const float* bq = (const float*)d_in[5];
  const float* wk = (const float*)d_in[6];
  const float* bk = (const float*)d_in[7];
  const float* wv = (const float*)d_in[8];
  const float* bv = (const float*)d_in[9];
  const float* wo = (const float*)d_in[10];
  const float* bo = (const float*)d_in[11];
  float* out = (float*)d_out;

  _Float16* ws = (_Float16*)d_ws;
  const size_t MW = (size_t)1024 * 1024;   // halves per 1024x1024 f16 matrix
  _Float16* wqT = ws + 0 * MW;
  _Float16* wkT = ws + 1 * MW;
  _Float16* wvT = ws + 2 * MW;
  _Float16* woT = ws + 3 * MW;
  _Float16* Qs  = ws + 4 * MW;    // [B,H,S,64] = 8 MW
  _Float16* Ks  = ws + 12 * MW;
  _Float16* Vs  = ws + 20 * MW;
  _Float16* Ao  = ws + 28 * MW;   // [B*S, 1024] = 8 MW

  dim3 blk(256);

  k_transpose_w<<<4096, blk, 0, stream>>>(wq, wqT);
  k_transpose_w<<<4096, blk, 0, stream>>>(wk, wkT);
  k_transpose_w<<<4096, blk, 0, stream>>>(wv, wvT);
  k_transpose_w<<<4096, blk, 0, stream>>>(wo, woT);

  dim3 g2(32, 16);   // 8192/256 x 1024/64
  k_qkv_gemm<<<g2, blk, 0, stream>>>(query, wqT, bq, Qs);
  k_qkv_gemm<<<g2, blk, 0, stream>>>(key,   wkT, bk, Ks);
  k_qkv_gemm<<<g2, blk, 0, stream>>>(value, wvT, bv, Vs);

  dim3 g3(16, 64);   // 2048/128 q-tiles x B*H
  k_attn<<<g3, blk, 0, stream>>>(Qs, Ks, Vs, amask, Ao);

  k_out_gemm<<<g2, blk, 0, stream>>>(Ao, woT, bo, out);
}
